// PackedLSTM_60627758350801
// MI455X (gfx1250) — matrix operand order repaired and checked by reading, not compiled, not run on hardware
//
#include <hip/hip_runtime.h>

// ---------------------------------------------------------------------------
// Bidirectional 2-layer LSTM for MI455X (gfx1250), wave32 + WMMA f16->f32.
//
// Phases (all on `stream`, serialized by stream order):
//   1. convert x, Wih, Whh to f16 (L2-resident working set)
//   2. lens[b] from mask
//   3. per layer: proj GEMM P = X@Wih^T + (bih+bhh)  (WMMA, time-parallel)
//   4. per layer: persistent scan kernel, 16 WGs/direction, atomic-counter
//      grid barrier per timestep with agent release/acquire fences;
//      recurrent GEMM h@Whh^T via v_wmma_f32_16x16x32_f16.
// ---------------------------------------------------------------------------

typedef _Float16 h8v  __attribute__((ext_vector_type(8)));
typedef _Float16 v16h __attribute__((ext_vector_type(16)));
typedef float    v8f  __attribute__((ext_vector_type(8)));

#define LSTM_T 512
#define LSTM_B 32
#define LSTM_I 2048
#define LSTM_H 1024
#define LSTM_N4 4096                 // 4*H
#define LSTM_M (LSTM_T * LSTM_B)     // 16384 rows in projection GEMM
#define SCAN_WGS 16                  // workgroups per direction in scan

// ---- WMMA tile loaders -----------------------------------------------------
// A (16x32 f16, M x K): lane<16 -> M=lane, halves 0..7 = K=kg+0..7,
// halves 8..15 = K=16+kg+0..7 with kg = (lane>=16 ? 8 : 0). Two 16B runs.
__device__ __forceinline__ v16h load_a16(const _Float16* __restrict__ p,
                                         int ld, int m0, int k0, int lane) {
  int m  = m0 + (lane & 15);
  int kg = (lane >> 4) << 3;
  const _Float16* r = p + (size_t)m * ld + (k0 + kg);
  union { v16h v; h8v h[2]; } u;
  u.h[0] = *(const h8v*)(r);
  u.h[1] = *(const h8v*)(r + 16);
  return u.v;
}

// B (32x16 f16, K x N), sourced from row-major W[n][k] (so B[k][n] = W[n][k]):
// lane gives N=lane&15; same K layout as A: K=kg+0..7 and K=16+kg+0..7, kg=(lane>=16?8:0).
__device__ __forceinline__ v16h load_b16(const _Float16* __restrict__ w,
                                         int ld, int n0, int k0, int lane) {
  int n  = n0 + (lane & 15);
  int ko = k0 + ((lane >> 4) << 3);
  const _Float16* r = w + (size_t)n * ld + ko;
  union { v16h v; h8v h[2]; } u;
  u.h[0] = *(const h8v*)(r);
  u.h[1] = *(const h8v*)(r + 16);
  return u.v;
}

__device__ __forceinline__ v8f wmma16(v16h a, v16h b, v8f c) {
  return __builtin_amdgcn_wmma_f32_16x16x32_f16(false, a, false, b,
                                                (short)0, c, false, false);
}

__device__ __forceinline__ float sigmoidf_(float x) {
  return 1.0f / (1.0f + __expf(-x));
}

// ---- utility kernels -------------------------------------------------------
__global__ void k_f32_to_f16(const float* __restrict__ in,
                             _Float16* __restrict__ out, size_t n) {
  size_t i = (size_t)blockIdx.x * blockDim.x + threadIdx.x;
  size_t s = (size_t)gridDim.x * blockDim.x;
  for (; i < n; i += s) out[i] = (_Float16)in[i];
}

__global__ void k_zero_u32(unsigned* __restrict__ p, size_t n) {
  size_t i = (size_t)blockIdx.x * blockDim.x + threadIdx.x;
  size_t s = (size_t)gridDim.x * blockDim.x;
  for (; i < n; i += s) p[i] = 0u;
}

// mask is a jnp bool array -> 1 byte per element, shape (B, T)
__global__ void k_lens(const unsigned char* __restrict__ mask,
                       int* __restrict__ lens) {
  int b = threadIdx.x;
  if (b < LSTM_B) {
    int cnt = 0;
    for (int t = 0; t < LSTM_T; ++t) cnt += (mask[(size_t)b * LSTM_T + t] != 0);
    lens[b] = LSTM_T - cnt;   // valid = t < len
  }
}

// ---- input projection GEMM: P[d] = X @ W[d]^T + (bih[d] + bhh[d]) ----------
// X: M x 2048 (f16), W: (per dir) 4096 x 2048 (f16), P: (per dir) M x 4096 f32
// Block = 256 threads = 8 waves (2 M x 4 N), wave tile 32x64, block 64x256.
__global__ __launch_bounds__(256) void k_proj_gemm(
    const _Float16* __restrict__ X,
    const _Float16* __restrict__ Wb,   // + d*4096*2048
    const float* __restrict__ bih,     // + d*4096
    const float* __restrict__ bhh,
    float* __restrict__ Pb)            // + d*M*4096
{
  const int K = LSTM_I;
  const int d = blockIdx.z;
  const _Float16* W = Wb + (size_t)d * LSTM_N4 * K;
  const float* bi = bih + (size_t)d * LSTM_N4;
  const float* bh = bhh + (size_t)d * LSTM_N4;
  float* P = Pb + (size_t)d * LSTM_M * LSTM_N4;

  int lane = threadIdx.x & 31;
  int wave = threadIdx.x >> 5;
  int wm = wave >> 2, wn = wave & 3;
  int m0 = blockIdx.x * 64 + wm * 32;
  int n0 = blockIdx.y * 256 + wn * 64;

  v8f acc[2][4] = {};
  for (int k0 = 0; k0 < K; k0 += 32) {
    // prefetch next K-slab of X (emits global_prefetch_b8)
    if (k0 + 64 < K)
      __builtin_prefetch(X + (size_t)(m0 + (lane & 15)) * K + k0 + 64, 0, 0);
    v16h a0 = load_a16(X, K, m0,      k0, lane);
    v16h a1 = load_a16(X, K, m0 + 16, k0, lane);
#pragma unroll
    for (int j = 0; j < 4; ++j) {
      v16h b = load_b16(W, K, n0 + 16 * j, k0, lane);
      acc[0][j] = wmma16(a0, b, acc[0][j]);
      acc[1][j] = wmma16(a1, b, acc[1][j]);
    }
  }
#pragma unroll
  for (int i = 0; i < 2; ++i) {
#pragma unroll
    for (int j = 0; j < 4; ++j) {
      int col = n0 + 16 * j + (lane & 15);
      float bias = bi[col] + bh[col];
      int rbase = m0 + 16 * i + ((lane >> 4) << 3);
#pragma unroll
      for (int r = 0; r < 8; ++r)
        P[(size_t)(rbase + r) * LSTM_N4 + col] = acc[i][j][r] + bias;
    }
  }
}

// ---- persistent recurrent scan ---------------------------------------------
// Grid (16, 2): blockIdx.y = direction, 16 WGs per direction; each WG owns 64
// h-columns (=> 256 gate columns). Per step: gates = h@Whh^T (WMMA) + P gather,
// LDS exchange across gates, cell update, h write (double-buffered f16),
// atomic-counter barrier across the direction's 16 WGs.
__global__ __launch_bounds__(256) void k_lstm_scan(
    const float* __restrict__ Pb,        // + d*T*B*4096
    const _Float16* __restrict__ WhhB,   // + d*4096*1024
    const int* __restrict__ lens,
    float* __restrict__ hstate,          // [2][B][H] f32 (own slice only)
    float* __restrict__ cstate,          // [2][B][H] f32
    _Float16* __restrict__ h16,          // [2 bufs][2 dirs][B][H] f16
    float* __restrict__ out32,           // nullable, (T,B,2H), col off d*H
    _Float16* __restrict__ out16,        // nullable, same layout
    float* __restrict__ hn,              // + d*B*H
    float* __restrict__ cn,              // + d*B*H
    int* __restrict__ cnt)               // [2]
{
  const int d = blockIdx.y;
  const int j = blockIdx.x;                 // 0..15: h-col slice
  const float* P = Pb + (size_t)d * LSTM_T * LSTM_B * LSTM_N4;
  const _Float16* Whh = WhhB + (size_t)d * LSTM_N4 * LSTM_H;
  float* hs = hstate + (size_t)d * LSTM_B * LSTM_H;
  float* cs = cstate + (size_t)d * LSTM_B * LSTM_H;
  int* cptr = cnt + d;

  int tid = threadIdx.x, lane = tid & 31, wave = tid >> 5;
  int g   = wave >> 1;                      // gate 0..3
  int sub = (wave & 1) * 32;                // 0 or 32 within 64-col slice
  int ncol0 = j * 64;                       // h-col base of this WG

  __shared__ float gbuf[4][64][33];         // [gate][n_local][b], padded
  __shared__ int slens[LSTM_B];
  if (tid < LSTM_B) slens[tid] = lens[tid];
  __syncthreads();

  for (int s = 0; s < LSTM_T; ++s) {
    const _Float16* h16r = h16 + ((size_t)(s & 1) * 2 + d) * LSTM_B * LSTM_H;
    _Float16* h16w = h16 + ((size_t)((s & 1) ^ 1) * 2 + d) * LSTM_B * LSTM_H;

    // recurrent GEMM: 32 (batch) x 32 (gate cols of this wave) x 1024
    v8f acc[2][2] = {};
    int nrow0 = g * LSTM_H + ncol0 + sub;   // Whh row base (32 rows)
    for (int k0 = 0; k0 < LSTM_H; k0 += 32) {
      v16h a0 = load_a16(h16r, LSTM_H, 0,  k0, lane);
      v16h a1 = load_a16(h16r, LSTM_H, 16, k0, lane);
      v16h b0 = load_b16(Whh, LSTM_H, nrow0,      k0, lane);
      v16h b1 = load_b16(Whh, LSTM_H, nrow0 + 16, k0, lane);
      acc[0][0] = wmma16(a0, b0, acc[0][0]);
      acc[1][0] = wmma16(a1, b0, acc[1][0]);
      acc[0][1] = wmma16(a0, b1, acc[0][1]);
      acc[1][1] = wmma16(a1, b1, acc[1][1]);
    }

    // add gathered P (per-batch time index) and exchange via LDS
#pragma unroll
    for (int i = 0; i < 2; ++i) {
#pragma unroll
      for (int jt = 0; jt < 2; ++jt) {
        int col_local = sub + 16 * jt + (lane & 15);       // 0..63
        int gcol = g * LSTM_H + ncol0 + col_local;
        int bbase = 16 * i + ((lane >> 4) << 3);
#pragma unroll
        for (int r = 0; r < 8; ++r) {
          int b = bbase + r;
          int len = slens[b];
          int tb = (s < len) ? (d ? (len - 1 - s) : s) : s;
          gbuf[g][col_local][b] =
              acc[i][jt][r] + P[((size_t)tb * LSTM_B + b) * LSTM_N4 + gcol];
        }
      }
    }
    __syncthreads();

    // cell update for the 32 x 64 (b, n_local) pairs of this WG
    for (int e = tid; e < LSTM_B * 64; e += 256) {
      int b  = e & 31;
      int nl = e >> 5;
      int len = slens[b];
      if (s < len) {
        float iv = sigmoidf_(gbuf[0][nl][b]);
        float fv = sigmoidf_(gbuf[1][nl][b]);
        float gv = tanhf(gbuf[2][nl][b]);
        float ov = sigmoidf_(gbuf[3][nl][b]);
        int n = ncol0 + nl;
        size_t hi = (size_t)b * LSTM_H + n;
        float c_new = fv * cs[hi] + iv * gv;
        float h_new = ov * tanhf(c_new);
        cs[hi] = c_new;
        hs[hi] = h_new;
        h16w[hi] = (_Float16)h_new;
        int tb = d ? (len - 1 - s) : s;
        size_t oi = ((size_t)tb * LSTM_B + b) * (2 * LSTM_H) + d * LSTM_H + n;
        if (out32) out32[oi] = h_new;
        if (out16) out16[oi] = (_Float16)h_new;
      }
      // invalid (masked) steps: h/c persist; stale rows in the write buffer
      // only affect gates of the same masked batch row -> discarded.
    }
    __syncthreads();

    // device-wide barrier across this direction's 16 WGs
    __builtin_amdgcn_fence(__ATOMIC_RELEASE, "agent");
    if (tid == 0) {
      __hip_atomic_fetch_add(cptr, 1, __ATOMIC_RELAXED, __HIP_MEMORY_SCOPE_AGENT);
      while (__hip_atomic_load(cptr, __ATOMIC_RELAXED,
                               __HIP_MEMORY_SCOPE_AGENT) < SCAN_WGS * (s + 1)) {
        __builtin_amdgcn_s_sleep(1);
      }
    }
    __syncthreads();
    __builtin_amdgcn_fence(__ATOMIC_ACQUIRE, "agent");
  }

  // final carries h_n / c_n for this WG's slice
  for (int e = tid; e < LSTM_B * 64; e += 256) {
    int b  = e & 31;
    int nl = e >> 5;
    int n = ncol0 + nl;
    size_t hi = (size_t)b * LSTM_H + n;
    size_t oi = (size_t)d * LSTM_B * LSTM_H + (size_t)b * LSTM_H + n;
    hn[oi] = hs[hi];
    cn[oi] = cs[hi];
  }
}

// ---------------------------------------------------------------------------
extern "C" void kernel_launch(void* const* d_in, const int* in_sizes, int n_in,
                              void* d_out, int out_size, void* d_ws, size_t ws_size,
                              hipStream_t stream) {
  const float* x   = (const float*)d_in[0];
  const unsigned char* mask = (const unsigned char*)d_in[1];  // jnp bool
  const float* Wih = (const float*)d_in[2];   // (2,2,4096,2048)
  const float* Whh = (const float*)d_in[3];   // (2,2,4096,1024)
  const float* bih = (const float*)d_in[4];   // (2,2,4096)
  const float* bhh = (const float*)d_in[5];
  float* out = (float*)d_out;

  // workspace layout (bytes)
  char* base = (char*)d_ws;
  const size_t OFF_P     = 0;                               // 536,870,912
  const size_t OFF_WIH16 = OFF_P     + (size_t)2 * LSTM_M * LSTM_N4 * 4;
  const size_t OFF_WHH16 = OFF_WIH16 + (size_t)4 * LSTM_N4 * LSTM_I * 2;
  const size_t OFF_X16   = OFF_WHH16 + (size_t)4 * LSTM_N4 * LSTM_H * 2;
  const size_t OFF_Y16   = OFF_X16   + (size_t)LSTM_M * LSTM_I * 2;
  const size_t OFF_H16   = OFF_Y16   + (size_t)LSTM_M * LSTM_I * 2;
  const size_t OFF_HS    = OFF_H16   + (size_t)2 * 2 * LSTM_B * LSTM_H * 2;
  const size_t OFF_CS    = OFF_HS    + (size_t)2 * LSTM_B * LSTM_H * 4;
  const size_t OFF_LENS  = OFF_CS    + (size_t)2 * LSTM_B * LSTM_H * 4;
  const size_t OFF_CNT   = OFF_LENS  + 128;

  float*    P     = (float*)(base + OFF_P);
  _Float16* Wih16 = (_Float16*)(base + OFF_WIH16);
  _Float16* Whh16 = (_Float16*)(base + OFF_WHH16);
  _Float16* X16   = (_Float16*)(base + OFF_X16);
  _Float16* Y16   = (_Float16*)(base + OFF_Y16);
  _Float16* H16   = (_Float16*)(base + OFF_H16);
  float*    HS    = (float*)(base + OFF_HS);
  float*    CS    = (float*)(base + OFF_CS);
  int*      LENS  = (int*)(base + OFF_LENS);
  int*      CNT   = (int*)(base + OFF_CNT);

  const size_t INP_WORDS = (size_t)LSTM_M * 2 * LSTM_H;     // final output (T,B,2H)
  float* HN = out + INP_WORDS;                              // (4,B,H)
  float* CN = HN + (size_t)4 * LSTM_B * LSTM_H;

  // 1. zero output inp region + layer-0 y (f16) buffer (masked slots stay 0)
  k_zero_u32<<<2048, 256, 0, stream>>>((unsigned*)out, INP_WORDS);
  k_zero_u32<<<2048, 256, 0, stream>>>((unsigned*)Y16,
                                       (size_t)LSTM_M * LSTM_I * 2 / 4);

  // 2. f32 -> f16 conversions (weights + x)
  k_f32_to_f16<<<2048, 256, 0, stream>>>(x, X16, (size_t)LSTM_M * LSTM_I);
  k_f32_to_f16<<<2048, 256, 0, stream>>>(Wih, Wih16, (size_t)4 * LSTM_N4 * LSTM_I);
  k_f32_to_f16<<<2048, 256, 0, stream>>>(Whh, Whh16, (size_t)4 * LSTM_N4 * LSTM_H);

  // 3. sequence lengths
  k_lens<<<1, 32, 0, stream>>>(mask, LENS);

  const size_t STATE_WORDS = (OFF_LENS - OFF_H16) / 4;      // H16+HS+CS

  // ---- layer 0 ----
  k_zero_u32<<<256, 256, 0, stream>>>((unsigned*)(base + OFF_H16), STATE_WORDS);
  k_zero_u32<<<1, 32, 0, stream>>>((unsigned*)CNT, 32);
  k_proj_gemm<<<dim3(LSTM_M / 64, LSTM_N4 / 256, 2), 256, 0, stream>>>(
      X16, Wih16, bih, bhh, P);
  k_lstm_scan<<<dim3(SCAN_WGS, 2), 256, 0, stream>>>(
      P, Whh16, LENS, HS, CS, H16,
      (float*)nullptr, Y16,                     // layer-0 ys -> f16 feed only
      HN + 0, CN + 0, CNT);

  // ---- layer 1 (input = concat of layer-0 dir outputs, already in Y16) ----
  k_zero_u32<<<256, 256, 0, stream>>>((unsigned*)(base + OFF_H16), STATE_WORDS);
  k_zero_u32<<<1, 32, 0, stream>>>((unsigned*)CNT, 32);
  k_proj_gemm<<<dim3(LSTM_M / 64, LSTM_N4 / 256, 2), 256, 0, stream>>>(
      Y16, Wih16 + (size_t)2 * LSTM_N4 * LSTM_I,
      bih + (size_t)2 * LSTM_N4, bhh + (size_t)2 * LSTM_N4, P);
  k_lstm_scan<<<dim3(SCAN_WGS, 2), 256, 0, stream>>>(
      P, Whh16 + (size_t)2 * LSTM_N4 * LSTM_H, LENS, HS, CS, H16,
      out, (_Float16*)nullptr,                  // layer-1 ys -> final output
      HN + (size_t)2 * LSTM_B * LSTM_H,
      CN + (size_t)2 * LSTM_B * LSTM_H, CNT);
}
